// FFJORD_1743756722867
// MI455X (gfx1250) — matrix operand order repaired and checked by reading, not compiled, not run on hardware
//
#include <hip/hip_runtime.h>
#include <hip/hip_bf16.h>
#include <math.h>

// Problem dims (match reference)
#define BSZ  2048
#define DDIM 128
#define CDIM 4
#define HIDD 512
#define DINR 133      // real input width [y(128), cond(4), t(1)]
#define DINP 160      // padded to multiple of 32 for WMMA K-loop
#define OUTW 129      // D + 1
#define KMAX 512      // largest K dimension (for LDS sizing)

typedef __attribute__((ext_vector_type(16))) __bf16 v16bf;
typedef __attribute__((ext_vector_type(8)))  __bf16 v8bf;
typedef __attribute__((ext_vector_type(8)))  float  v8f;
typedef __attribute__((ext_vector_type(4)))  int    v4i;

enum { EPI_PLAIN = 0, EPI_BIAS = 1, EPI_TANH = 2, EPI_DTANH = 3 };

// Async global->LDS staging (gfx1250). Fall back to manual ds stores when the
// builtins aren't available (e.g. host-side pass of the HIP compilation).
#if defined(__gfx1250__) && \
    __has_builtin(__builtin_amdgcn_global_load_async_to_lds_b128) && \
    __has_builtin(__builtin_amdgcn_s_wait_asynccnt)
#define USE_ASYNC_LDS 1
#else
#define USE_ASYNC_LDS 0
#endif

// ---------------------------------------------------------------------------
// WMMA GEMM: C[M,N] = A[M,K] (bf16 row-major, lda) x Bp (bf16 packed [N][K])
//  - block = 8 waves (256 thr), covers 128(M) x NSUB*16(N)
//  - block stages its B slab (NSUB*16 rows x K, contiguous) into LDS via the
//    async global->LDS path; WMMA B-fragments then come from ds_load_b128
//  - each wave: one 16-row M tile x NSUB 16-col N subtiles (NSUB accumulators
//    share one A fragment per K-step)
// Fragment K order is contiguous: lane half h holds K 16h..16h+15 in both A and B (wave32).
// ---------------------------------------------------------------------------
template<int EPI, int NSUB>
__global__ __launch_bounds__(256)
void gemm_wmma(const __bf16* __restrict__ A, int lda,
               const __bf16* __restrict__ Bp,      // packed [N][K], ldb == K
               const float*  __restrict__ bias,
               const __bf16* __restrict__ hbuf,    // for EPI_DTANH (same ld as C)
               float* __restrict__ Cf, __bf16* __restrict__ Cbf, int ldc,
               int K)
{
    __shared__ __bf16 ldsB[NSUB * 16 * KMAX];

    const int lane = threadIdx.x & 31;
    const int wave = threadIdx.x >> 5;
    const int m0 = (blockIdx.y * 8 + wave) * 16;
    const int n0 = blockIdx.x * (NSUB * 16);

    // ---- stage B slab: rows n0..n0+NSUB*16-1, contiguous since ldb == K ----
    {
        const __bf16* bsrc = Bp + (size_t)n0 * K;
        const int nchunks = (NSUB * 16 * K) / 8;      // 16-byte chunks
        for (int c = threadIdx.x; c < nchunks; c += 256) {
            const int e = c * 8;
#if USE_ASYNC_LDS
            __builtin_amdgcn_global_load_async_to_lds_b128(
                (__attribute__((address_space(1))) v4i*)(bsrc + e),
                (__attribute__((address_space(3))) v4i*)(ldsB + e),
                0, 0);
#else
            *(v8bf*)(ldsB + e) = *(const v8bf*)(bsrc + e);
#endif
        }
#if USE_ASYNC_LDS
        __builtin_amdgcn_s_wait_asynccnt(0);
#endif
        __syncthreads();
    }

    // ---- fragment addressing ----
    const int arow = m0 + (lane & 15);     // A: row, half-wave K offset 0/16
    const int aoff = (lane >> 4) * 16;
    const int blane = lane & 15;           // B: col-within-subtile, K off 0/16
    const int boff = (lane >> 4) * 16;

    const __bf16* ap = A + (size_t)arow * lda + aoff;

    v8f acc[NSUB];
#pragma unroll
    for (int j = 0; j < NSUB; ++j) acc[j] = v8f{0.f, 0.f, 0.f, 0.f, 0.f, 0.f, 0.f, 0.f};

    for (int k0 = 0; k0 < K; k0 += 32) {
        // A fragment: VGPR0-3 = K aoff+0..7, VGPR4-7 = K aoff+8..15
        v8bf alo = *(const v8bf*)(ap + k0);
        v8bf ahi = *(const v8bf*)(ap + k0 + 8);
        v16bf af;
#pragma unroll
        for (int i = 0; i < 8; ++i) { af[i] = alo[i]; af[8 + i] = ahi[i]; }
#pragma unroll
        for (int j = 0; j < NSUB; ++j) {
            // B fragment: 16 consecutive K for this lane's column, from LDS
            const __bf16* bp = ldsB + (size_t)(j * 16 + blane) * K + k0 + boff;
            v16bf bf = *(const v16bf*)bp;
            acc[j] = __builtin_amdgcn_wmma_f32_16x16x32_bf16(
                false, af, false, bf, (short)0, acc[j], false, false);
        }
    }

    // C/D layout: VGPR r -> row m0 + (lane>=16 ? 8 : 0) + r, col (lane&15)
    const int crow = m0 + (lane >> 4) * 8;
#pragma unroll
    for (int j = 0; j < NSUB; ++j) {
        const int ccol = n0 + j * 16 + (lane & 15);
#pragma unroll
        for (int r = 0; r < 8; ++r) {
            const size_t idx = (size_t)(crow + r) * ldc + ccol;
            float v = acc[j][r];
            if constexpr (EPI == EPI_PLAIN) {
                Cf[idx] = v;
            } else if constexpr (EPI == EPI_BIAS) {
                Cf[idx] = v + bias[ccol];
            } else if constexpr (EPI == EPI_TANH) {
                Cbf[idx] = (__bf16)tanhf(v + bias[ccol]);
            } else { // EPI_DTANH: grad * (1 - h^2)
                float h = (float)hbuf[idx];
                Cbf[idx] = (__bf16)(v * (1.0f - h * h));
            }
        }
    }
}

// ---------------------------------------------------------------------------
// Elementwise / packing kernels
// ---------------------------------------------------------------------------
__global__ void pack_tr(const float* __restrict__ W, __bf16* __restrict__ out,
                        int K, int N, int ldk)  // out[n*ldk + k] = W[k*N + n]
{
    int i = blockIdx.x * blockDim.x + threadIdx.x;
    if (i >= N * ldk) return;
    int n = i / ldk, k = i % ldk;
    out[i] = (__bf16)(k < K ? W[(size_t)k * N + n] : 0.f);
}

__global__ void pack_cp(const float* __restrict__ in, __bf16* __restrict__ out, int n)
{
    int i = blockIdx.x * blockDim.x + threadIdx.x;
    if (i < n) out[i] = (__bf16)in[i];
}

__global__ void build_input(const float* __restrict__ y, const float* __restrict__ cond,
                            float t, __bf16* __restrict__ hin)
{
    int i = blockIdx.x * blockDim.x + threadIdx.x;
    if (i >= BSZ * DINP) return;
    int m = i / DINP, c = i % DINP;
    float v;
    if (c < DDIM)             v = y[(size_t)m * DDIM + c];
    else if (c < DDIM + CDIM) v = cond[(size_t)m * CDIM + (c - DDIM)];
    else if (c == DDIM + CDIM) v = t;
    else                      v = 0.f;
    hin[i] = (__bf16)v;
}

__global__ void reduce_dld(const float* __restrict__ eps, const float* __restrict__ gx,
                           float* __restrict__ dld)
{
    int m = blockIdx.x * blockDim.x + threadIdx.x;
    if (m >= BSZ) return;
    float s = 0.f;
    const float* e = eps + (size_t)m * DDIM;
    const float* g = gx  + (size_t)m * DDIM;
#pragma unroll 8
    for (int j = 0; j < DDIM; ++j) s += e[j] * g[j];
    dld[m] = s;
}

__global__ void axpy_k(const float* __restrict__ a, const float* __restrict__ b,
                       float s, float* __restrict__ o, int n)
{
    int i = blockIdx.x * blockDim.x + threadIdx.x;
    if (i < n) o[i] = a[i] + s * b[i];
}

__global__ void combine_k(float* __restrict__ y,
                          const float* __restrict__ k1, const float* __restrict__ k2,
                          const float* __restrict__ k3, const float* __restrict__ k4,
                          float s, int n)
{
    int i = blockIdx.x * blockDim.x + threadIdx.x;
    if (i < n) y[i] += s * (k1[i] + 2.f * k2[i] + 2.f * k3[i] + k4[i]);
}

__global__ void write_out(const float* __restrict__ y, const float* __restrict__ ld,
                          float* __restrict__ out)
{
    int i = blockIdx.x * blockDim.x + threadIdx.x;
    if (i >= BSZ * OUTW) return;
    int m = i / OUTW, c = i % OUTW;
    out[i] = (c < DDIM) ? y[(size_t)m * DDIM + c] : ld[m];
}

// ---------------------------------------------------------------------------
// Host orchestration: fixed-step RK4 (24 steps) of the augmented ODE.
// ---------------------------------------------------------------------------
static inline void* carve(char*& p, size_t bytes)
{
    void* r = (void*)p;
    p += (bytes + 255) & ~(size_t)255;
    return r;
}

extern "C" void kernel_launch(void* const* d_in, const int* in_sizes, int n_in,
                              void* d_out, int out_size, void* d_ws, size_t ws_size,
                              hipStream_t stream)
{
    const float* x    = (const float*)d_in[0];
    const float* cond = (const float*)d_in[1];
    const float* eps  = (const float*)d_in[2];
    const float* W1   = (const float*)d_in[3];
    const float* b1   = (const float*)d_in[4];
    const float* W2   = (const float*)d_in[5];
    const float* b2   = (const float*)d_in[6];
    const float* W3   = (const float*)d_in[7];
    const float* b3   = (const float*)d_in[8];
    float* out = (float*)d_out;
    (void)in_sizes; (void)n_in; (void)out_size; (void)ws_size;

    char* p = (char*)d_ws;
    // bf16 packed weights. Forward: [N][K] transposed; backward: direct copies.
    __bf16* F1    = (__bf16*)carve(p, (size_t)HIDD * DINP * 2);  // [512][160]
    __bf16* F2    = (__bf16*)carve(p, (size_t)HIDD * HIDD * 2);  // [512][512]
    __bf16* F3    = (__bf16*)carve(p, (size_t)DDIM * HIDD * 2);  // [128][512]
    __bf16* B3    = (__bf16*)carve(p, (size_t)HIDD * DDIM * 2);  // = bf16(W3)
    __bf16* B2    = (__bf16*)carve(p, (size_t)HIDD * HIDD * 2);  // = bf16(W2)
    __bf16* B1    = (__bf16*)carve(p, (size_t)DDIM * HIDD * 2);  // = bf16(W1[:128,:])
    __bf16* epsbf = (__bf16*)carve(p, (size_t)BSZ * DDIM * 2);
    __bf16* hin   = (__bf16*)carve(p, (size_t)BSZ * DINP * 2);
    __bf16* h1bf  = (__bf16*)carve(p, (size_t)BSZ * HIDD * 2);
    __bf16* h2bf  = (__bf16*)carve(p, (size_t)BSZ * HIDD * 2);
    __bf16* g2bf  = (__bf16*)carve(p, (size_t)BSZ * HIDD * 2);
    __bf16* g1bf  = (__bf16*)carve(p, (size_t)BSZ * HIDD * 2);
    float*  gx    = (float*)carve(p, (size_t)BSZ * DDIM * 4);
    float*  y     = (float*)carve(p, (size_t)BSZ * DDIM * 4);
    float*  ytmp  = (float*)carve(p, (size_t)BSZ * DDIM * 4);
    float*  ld    = (float*)carve(p, (size_t)BSZ * 4);
    float*  kdy[4]; float* kld[4];
    for (int i = 0; i < 4; ++i) {
        kdy[i] = (float*)carve(p, (size_t)BSZ * DDIM * 4);
        kld[i] = (float*)carve(p, (size_t)BSZ * 4);
    }

    const int T = 256;
    auto blks = [](int n, int t) { return (n + t - 1) / t; };

    // --- pack weights (bf16) ---
    pack_tr<<<blks(HIDD * DINP, T), T, 0, stream>>>(W1, F1, DINR, HIDD, DINP);
    pack_tr<<<blks(HIDD * HIDD, T), T, 0, stream>>>(W2, F2, HIDD, HIDD, HIDD);
    pack_tr<<<blks(DDIM * HIDD, T), T, 0, stream>>>(W3, F3, HIDD, DDIM, HIDD);
    pack_cp<<<blks(HIDD * DDIM, T), T, 0, stream>>>(W3, B3, HIDD * DDIM);
    pack_cp<<<blks(HIDD * HIDD, T), T, 0, stream>>>(W2, B2, HIDD * HIDD);
    pack_cp<<<blks(DDIM * HIDD, T), T, 0, stream>>>(W1, B1, DDIM * HIDD);
    pack_cp<<<blks(BSZ * DDIM, T), T, 0, stream>>>(eps, epsbf, BSZ * DDIM);

    // --- init state ---
    (void)hipMemcpyAsync(y, x, (size_t)BSZ * DDIM * 4, hipMemcpyDeviceToDevice, stream);
    (void)hipMemsetAsync(ld, 0, (size_t)BSZ * 4, stream);

    const dim3 gH(HIDD / 64, BSZ / 128);   // N=512: NSUB=4 (16x64 per wave)
    const dim3 gD(DDIM / 32, BSZ / 128);   // N=128: NSUB=2 (16x32 per wave)
    const dim3 blk(256);

    // one augmented-dynamics eval: (dy, dld) = aug_dyn(t, yin)
    auto rhs = [&](float t, const float* yin, float* dy, float* dl) {
        build_input<<<blks(BSZ * DINP, T), T, 0, stream>>>(yin, cond, t, hin);
        // forward
        gemm_wmma<EPI_TANH , 4><<<gH, blk, 0, stream>>>(hin,  DINP, F1, b1, nullptr, nullptr, h1bf, HIDD, DINP);
        gemm_wmma<EPI_TANH , 4><<<gH, blk, 0, stream>>>(h1bf, HIDD, F2, b2, nullptr, nullptr, h2bf, HIDD, HIDD);
        gemm_wmma<EPI_BIAS , 2><<<gD, blk, 0, stream>>>(h2bf, HIDD, F3, b3, nullptr, dy, nullptr, DDIM, HIDD);
        // VJP: eps @ W3^T -> *(1-h2^2) -> @ W2^T -> *(1-h1^2) -> @ W1^T[:, :128]
        gemm_wmma<EPI_DTANH, 4><<<gH, blk, 0, stream>>>(epsbf, DDIM, B3, nullptr, h2bf, nullptr, g2bf, HIDD, DDIM);
        gemm_wmma<EPI_DTANH, 4><<<gH, blk, 0, stream>>>(g2bf,  HIDD, B2, nullptr, h1bf, nullptr, g1bf, HIDD, HIDD);
        gemm_wmma<EPI_PLAIN, 2><<<gD, blk, 0, stream>>>(g1bf,  HIDD, B1, nullptr, nullptr, gx, nullptr, DDIM, HIDD);
        reduce_dld<<<blks(BSZ, T), T, 0, stream>>>(eps, gx, dl);
    };

    const int NSTEP = 24;
    const float hs = 1.0f / (float)NSTEP;
    const int nyd = BSZ * DDIM;

    for (int s = 0; s < NSTEP; ++s) {
        float t0 = (float)s * hs;
        rhs(t0, y, kdy[0], kld[0]);
        axpy_k<<<blks(nyd, T), T, 0, stream>>>(y, kdy[0], 0.5f * hs, ytmp, nyd);
        rhs(t0 + 0.5f * hs, ytmp, kdy[1], kld[1]);
        axpy_k<<<blks(nyd, T), T, 0, stream>>>(y, kdy[1], 0.5f * hs, ytmp, nyd);
        rhs(t0 + 0.5f * hs, ytmp, kdy[2], kld[2]);
        axpy_k<<<blks(nyd, T), T, 0, stream>>>(y, kdy[2], hs, ytmp, nyd);
        rhs(t0 + hs, ytmp, kdy[3], kld[3]);
        combine_k<<<blks(nyd, T), T, 0, stream>>>(y, kdy[0], kdy[1], kdy[2], kdy[3], hs / 6.f, nyd);
        combine_k<<<blks(BSZ, T), T, 0, stream>>>(ld, kld[0], kld[1], kld[2], kld[3], hs / 6.f, BSZ);
    }

    write_out<<<blks(BSZ * OUTW, T), T, 0, stream>>>(y, ld, out);
}
